// RelationMessagePassing_52776558133695
// MI455X (gfx1250) — matrix operand order repaired and checked by reading, not compiled, not run on hardware
//
#include <hip/hip_runtime.h>

// MI455X / gfx1250, wave32. bf16 WMMA (16x16x32) with f32 accumulate.
// ~67 GFLOP of GEMM work; mandatory data movement ~1-1.5 GB -> ~50-70us at
// 23.3 TB/s, so GEMMs go through v_wmma_f32_16x16x32_bf16 (matrix pipes) and
// the scatter-add uses native global_atomic_add_f32 into an f32 accumulator.

typedef unsigned short u16;
typedef __attribute__((ext_vector_type(16))) __bf16 v16bf;
typedef __attribute__((ext_vector_type(8)))  float  v8f;

union FragB16 { uint4 u[2]; v16bf v; };

static __device__ __forceinline__ u16 f2bf(float f) {
  union { float f; unsigned u; } c; c.f = f;
  unsigned u = c.u;
  return (u16)((u + 0x7FFFu + ((u >> 16) & 1u)) >> 16);   // RNE
}

// Force the native f32 atomic (global_atomic_add_f32 at device scope).
static __device__ __forceinline__ void atomic_add_f32(float* p, float v) {
  __hip_atomic_fetch_add(p, v, __ATOMIC_RELAXED, __HIP_MEMORY_SCOPE_AGENT);
}

static constexpr int TM = 64;   // tuples (or nodes) per block

// ---------------- prep kernels ----------------------------------------------

__global__ void k_f32_to_bf16(const float* __restrict__ src,
                              u16* __restrict__ dst, int n) {
  int i = blockIdx.x * 256 + threadIdx.x;
  if (i < n) dst[i] = f2bf(src[i]);
}

// W is [din,dout] row-major (x @ W); Wt[n][k] = W[k][n], bf16.
__global__ void k_prep_wt(const float* __restrict__ W,
                          u16* __restrict__ Wt, int din, int dout) {
  int i = blockIdx.x * 256 + threadIdx.x;
  if (i < din * dout) {
    int k = i / dout, n = i % dout;
    Wt[(size_t)n * din + k] = f2bf(W[i]);
  }
}

// ---------------- relation MLP + scatter ------------------------------------
// msg = relu(X @ W1 + b1) @ W2 + b2 ; scatter-add rows (reshaped [T*a,64]).

template <int D, int ARITY>
__launch_bounds__(256)
__global__ void k_relation(const u16* __restrict__ ns_bf,
                           const long long* __restrict__ idx,
                           const u16* __restrict__ Wt1,
                           const float* __restrict__ b1,
                           const u16* __restrict__ Wt2,
                           const float* __restrict__ b2,
                           float* __restrict__ sum_msg,
                           int T) {
  constexpr int NT = D / 16;
  constexpr int KT = D / 32;
  constexpr int LD = D + 8;        // +8 u16: 16B-aligned rows, no bank clash
  __shared__ u16 Xs[TM][LD];
  __shared__ u16 Hs[TM][LD];

  const int tid = threadIdx.x;
  const int t0  = blockIdx.x * TM;

  // ---- gather X tile: row m = tuple t0+m, D cols = concat of node states.
  // 16B chunks (8 bf16) never straddle a 64-col node segment (8 | 64).
  constexpr int CH = D / 8;
  for (int e = tid; e < TM * CH; e += 256) {
    const int m  = e / CH;
    const int c8 = (e % CH) * 8;
    uint4 val; val.x = val.y = val.z = val.w = 0u;
    const int t = t0 + m;
    if (t < T) {
      const long long node = idx[(long long)t * ARITY + (c8 >> 6)];
      val = *(const uint4*)(ns_bf + node * 64 + (c8 & 63));
    }
    *(uint4*)&Xs[m][c8] = val;
  }
  __syncthreads();

  const int w    = tid >> 5;
  const int lane = tid & 31;
  const int mt   = w >> 1;        // m-tile 0..3
  const int par  = w & 1;         // n-tile parity
  const int half = lane >> 4;
  const int ln   = lane & 15;
  const int arow = mt * 16 + ln;

  FragB16 a[KT];

  // ---- GEMM1: H = relu(X @ W1 + b1)
  // A layout (16-bit 16x32): elem e -> K = (e/8)*16 + (e%8) + half*8
#pragma unroll
  for (int kt = 0; kt < KT; ++kt) {
    a[kt].u[0] = *(const uint4*)&Xs[arow][kt * 32 + half * 8];
    a[kt].u[1] = *(const uint4*)&Xs[arow][kt * 32 + 16 + half * 8];
  }
#pragma unroll
  for (int nt = par; nt < NT; nt += 2) {
    v8f acc = {};
#pragma unroll
    for (int kt = 0; kt < KT; ++kt) {
      FragB16 b;   // B layout: lane n=ln, K = (e/8)*16 + (e%8) + half*8
      const uint4* wp =
          (const uint4*)(Wt1 + (size_t)(nt * 16 + ln) * D + kt * 32 + half * 8);
      b.u[0] = wp[0];
      b.u[1] = wp[2];
      acc = __builtin_amdgcn_wmma_f32_16x16x32_bf16(
          false, a[kt].v, false, b.v, (short)0, acc, false, false);
    }
    const float bias = b1[nt * 16 + ln];
#pragma unroll
    for (int v = 0; v < 8; ++v) {   // C/D: VGPR v -> M = v + 8*half, N = ln
      float h = acc[v] + bias;
      h = h > 0.0f ? h : 0.0f;
      Hs[mt * 16 + v + 8 * half][nt * 16 + ln] = f2bf(h);
    }
  }
  __syncthreads();

  // ---- GEMM2: MSG = H @ W2 + b2 ; scatter-add to sum_msg
#pragma unroll
  for (int kt = 0; kt < KT; ++kt) {
    a[kt].u[0] = *(const uint4*)&Hs[arow][kt * 32 + half * 8];
    a[kt].u[1] = *(const uint4*)&Hs[arow][kt * 32 + 16 + half * 8];
  }
#pragma unroll
  for (int nt = par; nt < NT; nt += 2) {
    v8f acc = {};
#pragma unroll
    for (int kt = 0; kt < KT; ++kt) {
      FragB16 b;
      const uint4* wp =
          (const uint4*)(Wt2 + (size_t)(nt * 16 + ln) * D + kt * 32 + half * 8);
      b.u[0] = wp[0];
      b.u[1] = wp[2];
      acc = __builtin_amdgcn_wmma_f32_16x16x32_bf16(
          false, a[kt].v, false, b.v, (short)0, acc, false, false);
    }
    const float bias = b2[nt * 16 + ln];
    const int part = nt >> 2;                 // which arity slot (16|64)
    const int col  = ((nt & 3) * 16) + ln;    // column within H=64
#pragma unroll
    for (int v = 0; v < 8; ++v) {
      const int t = t0 + mt * 16 + v + 8 * half;
      if (t < T) {
        const long long node = idx[(long long)t * ARITY + part];
        atomic_add_f32(sum_msg + node * 64 + col, acc[v] + bias);
      }
    }
  }
}

// ---------------- update MLP ------------------------------------------------
// out = relu([sum_msg | ns] @ wu1 + bu1) @ wu2 + bu2

__launch_bounds__(256)
__global__ void k_update(const float* __restrict__ sum_msg,
                         const u16* __restrict__ ns_bf,
                         const u16* __restrict__ Wtu1,   // [128][128]
                         const float* __restrict__ bu1,
                         const u16* __restrict__ Wtu2,   // [64][128]
                         const float* __restrict__ bu2,
                         float* __restrict__ out,
                         int Nn) {
  constexpr int D = 128, NT1 = 8, KT = 4, NT2 = 4, LD = D + 8;
  __shared__ u16 Zs[TM][LD];
  __shared__ u16 Hs[TM][LD];

  const int tid = threadIdx.x;
  const int n0  = blockIdx.x * TM;

  // stage z = [sum_msg | node_states], 8-col chunks
  constexpr int CH = D / 8;   // 16
  for (int e = tid; e < TM * CH; e += 256) {
    const int m  = e / CH;
    const int c8 = (e % CH) * 8;
    const int node = n0 + m;
    uint4 val; val.x = val.y = val.z = val.w = 0u;
    if (node < Nn) {
      if (c8 < 64) {
        const float4* sp = (const float4*)(sum_msg + (size_t)node * 64 + c8);
        float4 f0 = sp[0], f1 = sp[1];
        u16* q = (u16*)&val;
        q[0] = f2bf(f0.x); q[1] = f2bf(f0.y); q[2] = f2bf(f0.z); q[3] = f2bf(f0.w);
        q[4] = f2bf(f1.x); q[5] = f2bf(f1.y); q[6] = f2bf(f1.z); q[7] = f2bf(f1.w);
      } else {
        val = *(const uint4*)(ns_bf + (size_t)node * 64 + (c8 - 64));
      }
    }
    *(uint4*)&Zs[m][c8] = val;
  }
  __syncthreads();

  const int w    = tid >> 5;
  const int lane = tid & 31;
  const int mt   = w >> 1;
  const int par  = w & 1;
  const int half = lane >> 4;
  const int ln   = lane & 15;
  const int arow = mt * 16 + ln;

  FragB16 a[KT];
#pragma unroll
  for (int kt = 0; kt < KT; ++kt) {
    a[kt].u[0] = *(const uint4*)&Zs[arow][kt * 32 + half * 8];
    a[kt].u[1] = *(const uint4*)&Zs[arow][kt * 32 + 16 + half * 8];
  }
#pragma unroll
  for (int nt = par; nt < NT1; nt += 2) {
    v8f acc = {};
#pragma unroll
    for (int kt = 0; kt < KT; ++kt) {
      FragB16 b;
      const uint4* wp =
          (const uint4*)(Wtu1 + (size_t)(nt * 16 + ln) * D + kt * 32 + half * 8);
      b.u[0] = wp[0];
      b.u[1] = wp[2];
      acc = __builtin_amdgcn_wmma_f32_16x16x32_bf16(
          false, a[kt].v, false, b.v, (short)0, acc, false, false);
    }
    const float bias = bu1[nt * 16 + ln];
#pragma unroll
    for (int v = 0; v < 8; ++v) {
      float h = acc[v] + bias;
      h = h > 0.0f ? h : 0.0f;
      Hs[mt * 16 + v + 8 * half][nt * 16 + ln] = f2bf(h);
    }
  }
  __syncthreads();

#pragma unroll
  for (int kt = 0; kt < KT; ++kt) {
    a[kt].u[0] = *(const uint4*)&Hs[arow][kt * 32 + half * 8];
    a[kt].u[1] = *(const uint4*)&Hs[arow][kt * 32 + 16 + half * 8];
  }
#pragma unroll
  for (int nt = par; nt < NT2; nt += 2) {
    v8f acc = {};
#pragma unroll
    for (int kt = 0; kt < KT; ++kt) {
      FragB16 b;
      const uint4* wp =
          (const uint4*)(Wtu2 + (size_t)(nt * 16 + ln) * D + kt * 32 + half * 8);
      b.u[0] = wp[0];
      b.u[1] = wp[2];
      acc = __builtin_amdgcn_wmma_f32_16x16x32_bf16(
          false, a[kt].v, false, b.v, (short)0, acc, false, false);
    }
    const float bias = bu2[nt * 16 + ln];
#pragma unroll
    for (int v = 0; v < 8; ++v) {
      const int node = n0 + mt * 16 + v + 8 * half;
      if (node < Nn)
        out[(size_t)node * 64 + nt * 16 + ln] = acc[v] + bias;
    }
  }
}

// ---------------- host side -------------------------------------------------

extern "C" void kernel_launch(void* const* d_in, const int* in_sizes, int n_in,
                              void* d_out, int out_size, void* d_ws, size_t ws_size,
                              hipStream_t stream) {
  // setup_inputs() dict order:
  const float*     node_states = (const float*)d_in[0];
  const long long* idx0        = (const long long*)d_in[1];
  const float*     w1_r0 = (const float*)d_in[2];
  const float*     b1_r0 = (const float*)d_in[3];
  const float*     w2_r0 = (const float*)d_in[4];
  const float*     b2_r0 = (const float*)d_in[5];
  const long long* idx1        = (const long long*)d_in[6];
  const float*     w1_r1 = (const float*)d_in[7];
  const float*     b1_r1 = (const float*)d_in[8];
  const float*     w2_r1 = (const float*)d_in[9];
  const float*     b2_r1 = (const float*)d_in[10];
  const long long* idx2        = (const long long*)d_in[11];
  const float*     w1_r2 = (const float*)d_in[12];
  const float*     b1_r2 = (const float*)d_in[13];
  const float*     w2_r2 = (const float*)d_in[14];
  const float*     b2_r2 = (const float*)d_in[15];
  const float*     wu1   = (const float*)d_in[16];
  const float*     bu1   = (const float*)d_in[17];
  const float*     wu2   = (const float*)d_in[18];
  const float*     bu2   = (const float*)d_in[19];

  const int N  = in_sizes[0] / 64;
  const int T0 = in_sizes[1];        // arity 1
  const int T1 = in_sizes[6] / 2;    // arity 2
  const int T2 = in_sizes[11] / 3;   // arity 3

  // workspace carve-out (~77 MB)
  char*  ws  = (char*)d_ws;
  size_t off = 0;
  auto carve = [&](size_t bytes) {
    void* p = ws + off;
    off = (off + bytes + 255) & ~(size_t)255;
    return p;
  };
  float* sum_msg = (float*)carve((size_t)N * 64 * 4);
  u16*   ns_bf   = (u16*)carve((size_t)N * 64 * 2);
  u16*   wt1r0   = (u16*)carve((size_t)64 * 64 * 2);
  u16*   wt2r0   = (u16*)carve((size_t)64 * 64 * 2);
  u16*   wt1r1   = (u16*)carve((size_t)128 * 128 * 2);
  u16*   wt2r1   = (u16*)carve((size_t)128 * 128 * 2);
  u16*   wt1r2   = (u16*)carve((size_t)192 * 192 * 2);
  u16*   wt2r2   = (u16*)carve((size_t)192 * 192 * 2);
  u16*   wtu1    = (u16*)carve((size_t)128 * 128 * 2);
  u16*   wtu2    = (u16*)carve((size_t)64 * 128 * 2);
  (void)ws_size; (void)n_in; (void)out_size;

  // prep: bf16 node states + transposed bf16 weights, zero accumulator
  {
    int n = N * 64;
    k_f32_to_bf16<<<(n + 255) / 256, 256, 0, stream>>>(node_states, ns_bf, n);
  }
  auto prep = [&](const float* W, u16* Wt, int din, int dout) {
    int n = din * dout;
    k_prep_wt<<<(n + 255) / 256, 256, 0, stream>>>(W, Wt, din, dout);
  };
  prep(w1_r0, wt1r0, 64, 64);
  prep(w2_r0, wt2r0, 64, 64);
  prep(w1_r1, wt1r1, 128, 128);
  prep(w2_r1, wt2r1, 128, 128);
  prep(w1_r2, wt1r2, 192, 192);
  prep(w2_r2, wt2r2, 192, 192);
  prep(wu1,   wtu1,  128, 128);
  prep(wu2,   wtu2,  128, 64);
  hipMemsetAsync(sum_msg, 0, (size_t)N * 64 * 4, stream);

  // relation message passing (fused gather + 2 GEMMs + scatter-add)
  k_relation<64, 1><<<(T0 + TM - 1) / TM, 256, 0, stream>>>(
      ns_bf, idx0, wt1r0, b1_r0, wt2r0, b2_r0, sum_msg, T0);
  k_relation<128, 2><<<(T1 + TM - 1) / TM, 256, 0, stream>>>(
      ns_bf, idx1, wt1r1, b1_r1, wt2r1, b2_r1, sum_msg, T1);
  k_relation<192, 3><<<(T2 + TM - 1) / TM, 256, 0, stream>>>(
      ns_bf, idx2, wt1r2, b1_r2, wt2r2, b2_r2, sum_msg, T2);

  // final update MLP
  k_update<<<(N + TM - 1) / TM, 256, 0, stream>>>(
      sum_msg, ns_bf, wtu1, bu1, wtu2, bu2, (float*)d_out, N);
}
